// LSTMModel_70085276336931
// MI455X (gfx1250) — matrix operand order repaired and checked by reading, not compiled, not run on hardware
//
#include <hip/hip_runtime.h>

// ---------------------------------------------------------------------------
// Seq2seq LSTM (B=64, S=2048, H=512, L=2) for gfx1250 (MI455X).
// Persistent WMMA kernel: bf16 weights packed to WMMA-B fragment layout,
// recurrent GEMM on v_wmma_f32_16x16x32_bf16, c-state resident in LDS,
// h-state ping-pong in L2, grid barrier per layer step.
// ---------------------------------------------------------------------------

typedef __bf16 bf16;
typedef __attribute__((ext_vector_type(16))) bf16  v16bf;
typedef __attribute__((ext_vector_type(8)))  bf16  v8bf;
typedef __attribute__((ext_vector_type(8)))  float v8f;

#define BSZ   64
#define HSZ   512
#define SSZ   2048
#define NWG   16
#define TPB   512   // 16 waves of 32

// ---------------- helpers ----------------
__device__ __forceinline__ unsigned short f2bf_bits(float f) {
  unsigned u = __float_as_uint(f);
  u += 0x7FFFu + ((u >> 16) & 1u);          // round-to-nearest-even
  return (unsigned short)(u >> 16);
}
__device__ __forceinline__ float sigm(float v) { return 1.0f / (1.0f + expf(-v)); }

// ---------------- workspace zero ----------------
__global__ void ws_zero(uint4* p, int n) {
  int i = blockIdx.x * blockDim.x + threadIdx.x;
  if (i < n) { uint4 z; z.x = z.y = z.z = z.w = 0u; p[i] = z; }
}

// ---------------- weight packing: f32 [4H, K] -> bf16 fragment-major -------
// Fragment layout consumed by mm_frag: dst[((k>>5)*128+(g>>4))*512 + ((g&15)|((k&8)<<1))*16 + ((k&7)|((k&16)>>1))]
// i.e. lane L holds N column L&15, 16 k values in ISA order (h=L>>4: 8h+0..7, 16+8h+0..7).
__global__ void pack_weights(const float* __restrict__ eWhh0,
                             const float* __restrict__ eWih1,
                             const float* __restrict__ eWhh1,
                             const float* __restrict__ dWhh0,
                             const float* __restrict__ dWih1,
                             const float* __restrict__ dWhh1,
                             unsigned short* __restrict__ wL0e,
                             unsigned short* __restrict__ wL1e,
                             unsigned short* __restrict__ wL0d,
                             unsigned short* __restrict__ wL1d) {
  const int total = 3072 * 2048;     // k-rows: 512 + 1024 + 512 + 1024
  for (int idx = blockIdx.x * blockDim.x + threadIdx.x; idx < total;
       idx += gridDim.x * blockDim.x) {
    int kk = idx >> 11;              // global k-row among 3072
    int g  = idx & 2047;             // gate column 0..2047
    int k; float v; unsigned short* dst;
    if (kk < 512) {                  // encoder layer0: Whh0
      k = kk;         v = eWhh0[g * 512 + k];                                dst = wL0e;
    } else if (kk < 1536) {          // encoder layer1: [Wih1 ; Whh1]
      k = kk - 512;   v = (k < 512) ? eWih1[g * 512 + k] : eWhh1[g * 512 + (k - 512)];
      dst = wL1e;
    } else if (kk < 2048) {          // decoder layer0: Whh0
      k = kk - 1536;  v = dWhh0[g * 512 + k];                                dst = wL0d;
    } else {                         // decoder layer1: [Wih1 ; Whh1]
      k = kk - 2048;  v = (k < 512) ? dWih1[g * 512 + k] : dWhh1[g * 512 + (k - 512)];
      dst = wL1d;
    }
    dst[(((k >> 5) * 128 + (g >> 4)) * 32 + (g & 15) + (((k >> 3) & 1) << 4)) * 16 + (k & 7) + (((k >> 4) & 1) << 3)] = f2bf_bits(v);
  }
}

// ---------------- WMMA tile accumulation ----------------
// A: bf16 [64, 512] row-major (stride 512). B: packed fragments (see above).
// Computes two adjacent 16x16 output tiles (nt0, nt0+1) for row-tile mt,
// accumulating kt in [ktW0, ktW0+nKT) of the packed weight matrix.
// NOTE: unroll is capped at 2 -- full unroll makes the scheduler hoist all
// 32 B-fragment loads (256 VGPRs of data), spilling weights to scratch.
__device__ __forceinline__ void mm_frag(const bf16* __restrict__ Xb,
                                        const bf16* __restrict__ Wp,
                                        int ktW0, int nKT, int nt0, int mt, int lane,
                                        v8f& acc0, v8f& acc1) {
  const int row  = mt * 16 + (lane & 15);
  const int koff = (lane >> 4) << 3;             // ISA 16-bit A layout: lane-half K offset
  const bf16* xr = Xb + row * 512;
#pragma unroll 2
  for (int kt = 0; kt < nKT; ++kt) {
    union { v16bf v; v8bf h[2]; } a;
    const int kb = kt * 32 + koff;
    a.h[0] = *(const v8bf*)(xr + kb);            // K = base .. base+7
    a.h[1] = *(const v8bf*)(xr + kb + 16);       // K = base+16 .. base+23
    const bf16* wb = Wp + ((size_t)((ktW0 + kt) * 128 + nt0) * 32 + lane) * 16;
    v16bf b0 = *(const v16bf*)(wb);
    v16bf b1 = *(const v16bf*)(wb + 512);        // next n-tile (+32*16 elems)
    acc0 = __builtin_amdgcn_wmma_f32_16x16x32_bf16(false, a.v, false, b0,
                                                   (short)0, acc0, false, false);
    acc1 = __builtin_amdgcn_wmma_f32_16x16x32_bf16(false, a.v, false, b1,
                                                   (short)0, acc1, false, false);
  }
}

// ---------------- grid barrier (monotonic, 16 resident WGs) ----------------
__device__ __forceinline__ void grid_barrier(int* cnt, int expected) {
  __syncthreads();
  __threadfence();
  if (threadIdx.x == 0) {
    __hip_atomic_fetch_add(cnt, 1, __ATOMIC_RELEASE, __HIP_MEMORY_SCOPE_AGENT);
    while (__hip_atomic_load(cnt, __ATOMIC_ACQUIRE, __HIP_MEMORY_SCOPE_AGENT) <
           expected * NWG) {
      __builtin_amdgcn_s_sleep(2);
    }
  }
  __syncthreads();
}

// ---------------- one LSTM layer step (GEMM + gate fuse) ----------------
// A0 covers k 0..511; optional A1 covers k 512..1023 (stacked [x1 ; h_prev]).
__device__ __forceinline__ void do_layer(const bf16* __restrict__ A0,
                                         const bf16* __restrict__ A1,
                                         const bf16* __restrict__ Wp,
                                         const float* __restrict__ bih,
                                         const float* __restrict__ bhh,
                                         const float* __restrict__ wx,   // null for layer1
                                         const float* __restrict__ xv,   // per-batch scalar input
                                         int xvs,
                                         unsigned short* __restrict__ Hout,
                                         float* cbuf, float* h1b, float* gates,
                                         int tid, int lane, int mt, int q, int jbase) {
  v8f acc0 = {}; v8f acc1 = {};
  const int nt0 = (q * HSZ + jbase) >> 4;
  mm_frag(A0, Wp, 0, 16, nt0, mt, lane, acc0, acc1);
  if (A1) mm_frag(A1, Wp, 16, 16, nt0, mt, lane, acc0, acc1);

  // C/D layout: lane<16 -> (m = mt*16+r,   n = lane); lane>=16 -> (m = mt*16+8+r, n = lane-16)
  const int n  = lane & 15;
  const int mb = mt * 16 + ((lane >> 4) << 3);
#pragma unroll
  for (int r = 0; r < 8; ++r) {
    gates[(q * 64 + mb + r) * 32 + n]      = acc0[r];
    gates[(q * 64 + mb + r) * 32 + 16 + n] = acc1[r];
  }
  __syncthreads();

#pragma unroll
  for (int e = tid; e < 64 * 32; e += TPB) {
    const int b = e >> 5, j = e & 31;
    const int gcol = jbase + j;
    float pi = gates[(0 * 64 + b) * 32 + j] + bih[gcol]        + bhh[gcol];
    float pf = gates[(1 * 64 + b) * 32 + j] + bih[512  + gcol] + bhh[512  + gcol];
    float pg = gates[(2 * 64 + b) * 32 + j] + bih[1024 + gcol] + bhh[1024 + gcol];
    float po = gates[(3 * 64 + b) * 32 + j] + bih[1536 + gcol] + bhh[1536 + gcol];
    if (wx) {
      const float xb = xv[b * xvs];
      pi += xb * wx[gcol];        pf += xb * wx[512 + gcol];
      pg += xb * wx[1024 + gcol]; po += xb * wx[1536 + gcol];
    }
    const float cold = cbuf[e];
    const float cnew = sigm(pf) * cold + sigm(pi) * tanhf(pg);
    cbuf[e] = cnew;
    const float h = sigm(po) * tanhf(cnew);
    Hout[b * HSZ + gcol] = f2bf_bits(h);
    if (h1b) h1b[e] = h;
  }
}

// ---------------- persistent seq2seq kernel ----------------
__global__ void __launch_bounds__(TPB, 1)
lstm_seq2seq(const float* __restrict__ x,
             const float* __restrict__ ebih0, const float* __restrict__ ebhh0,
             const float* __restrict__ ewx,
             const float* __restrict__ ebih1, const float* __restrict__ ebhh1,
             const float* __restrict__ dbih0, const float* __restrict__ dbhh0,
             const float* __restrict__ dwx,
             const float* __restrict__ dbih1, const float* __restrict__ dbhh1,
             const float* __restrict__ fcW, const float* __restrict__ fcb,
             const unsigned short* __restrict__ WeL0, const unsigned short* __restrict__ WeL1,
             const unsigned short* __restrict__ WdL0, const unsigned short* __restrict__ WdL1,
             unsigned short* __restrict__ H0u, unsigned short* __restrict__ H1u,
             float* __restrict__ yp, int* cnt, float* __restrict__ out) {
  const int tid  = threadIdx.x;
  const int lane = tid & 31;
  const int ww   = tid >> 5;
  const int mt   = ww & 3;         // batch row-tile (64 rows = 4 tiles)
  const int q    = ww >> 2;        // gate index: 0=i 1=f 2=g 3=o
  const int jbase = blockIdx.x * 32;   // this WG's hidden-unit slice

  __shared__ float gates[4 * 64 * 32];   // 32 KB
  __shared__ float c0b[64 * 32];         // 8 KB  (layer0 cell state slice)
  __shared__ float c1b[64 * 32];         // 8 KB  (layer1 cell state slice)
  __shared__ float h1b[64 * 32];         // 8 KB  (layer1 h for fc partial)
  __shared__ float ylds[64];

  for (int e = tid; e < 64 * 32; e += TPB) { c0b[e] = 0.f; c1b[e] = 0.f; }
  __syncthreads();

  const bf16* WL0e = (const bf16*)WeL0;
  const bf16* WL1e = (const bf16*)WeL1;
  const bf16* WL0d = (const bf16*)WdL0;
  const bf16* WL1d = (const bf16*)WdL1;

  int phase = 0;

  // ================= encoder =================
  for (int t = 0; t < SSZ; ++t) {
    const int p = t & 1;
    const bf16* h0r = (const bf16*)H0u + p * (BSZ * HSZ);
    unsigned short* h0w = H0u + (p ^ 1) * (BSZ * HSZ);
    const bf16* h1r = (const bf16*)H1u + p * (BSZ * HSZ);
    unsigned short* h1w = H1u + (p ^ 1) * (BSZ * HSZ);

    // layer 0: g = h0_{t-1} @ Whh0^T + x_t * Wih0 + b
    do_layer(h0r, nullptr, WL0e, ebih0, ebhh0, ewx, x + t, SSZ,
             h0w, c0b, nullptr, gates, tid, lane, mt, q, jbase);
    grid_barrier(cnt, ++phase);

    // layer 1: g = [h0_t ; h1_{t-1}] @ [Wih1 ; Whh1]^T + b
    do_layer((const bf16*)h0w, h1r, WL1e, ebih1, ebhh1, nullptr, nullptr, 0,
             h1w, c1b, nullptr, gates, tid, lane, mt, q, jbase);
    grid_barrier(cnt, ++phase);
  }

  // ================= decoder (autoregressive) =================
  for (int t = 0; t < SSZ; ++t) {
    const int p = t & 1;
    const bf16* h0r = (const bf16*)H0u + p * (BSZ * HSZ);
    unsigned short* h0w = H0u + (p ^ 1) * (BSZ * HSZ);
    const bf16* h1r = (const bf16*)H1u + p * (BSZ * HSZ);
    unsigned short* h1w = H1u + (p ^ 1) * (BSZ * HSZ);

    // assemble y_{t-1} = fc_b + sum of the 16 deterministic partials
    if (tid < 64) {
      float y = 0.0f;
      if (t > 0) {
        y = fcb[0];
#pragma unroll
        for (int w = 0; w < NWG; ++w) y += yp[w * 64 + tid];
        if (blockIdx.x == 0) out[tid * SSZ + (t - 1)] = y;
      }
      ylds[tid] = y;
    }
    __syncthreads();

    do_layer(h0r, nullptr, WL0d, dbih0, dbhh0, dwx, ylds, 1,
             h0w, c0b, nullptr, gates, tid, lane, mt, q, jbase);
    grid_barrier(cnt, ++phase);

    do_layer((const bf16*)h0w, h1r, WL1d, dbih1, dbhh1, nullptr, nullptr, 0,
             h1w, c1b, h1b, gates, tid, lane, mt, q, jbase);
    __syncthreads();
    if (tid < 64) {                         // fc partial over this WG's 32 hidden units
      float s = 0.0f;
#pragma unroll
      for (int j = 0; j < 32; ++j) s += h1b[tid * 32 + j] * fcW[jbase + j];
      yp[blockIdx.x * 64 + tid] = s;
    }
    grid_barrier(cnt, ++phase);
  }

  // final output column t = S-1
  if (blockIdx.x == 0 && tid < 64) {
    float y = fcb[0];
#pragma unroll
    for (int w = 0; w < NWG; ++w) y += yp[w * 64 + tid];
    out[tid * SSZ + (SSZ - 1)] = y;
  }
}

// ---------------------------------------------------------------------------
extern "C" void kernel_launch(void* const* d_in, const int* in_sizes, int n_in,
                              void* d_out, int out_size, void* d_ws, size_t ws_size,
                              hipStream_t stream) {
  (void)in_sizes; (void)n_in; (void)out_size; (void)ws_size;

  const float* x     = (const float*)d_in[0];
  const float* eWih0 = (const float*)d_in[1];
  const float* eWhh0 = (const float*)d_in[2];
  const float* ebih0 = (const float*)d_in[3];
  const float* ebhh0 = (const float*)d_in[4];
  const float* eWih1 = (const float*)d_in[5];
  const float* eWhh1 = (const float*)d_in[6];
  const float* ebih1 = (const float*)d_in[7];
  const float* ebhh1 = (const float*)d_in[8];
  const float* dWih0 = (const float*)d_in[9];
  const float* dWhh0 = (const float*)d_in[10];
  const float* dbih0 = (const float*)d_in[11];
  const float* dbhh0 = (const float*)d_in[12];
  const float* dWih1 = (const float*)d_in[13];
  const float* dWhh1 = (const float*)d_in[14];
  const float* dbih1 = (const float*)d_in[15];
  const float* dbhh1 = (const float*)d_in[16];
  const float* fcW   = (const float*)d_in[17];
  const float* fcb   = (const float*)d_in[18];

  // ---- workspace layout (all 256-aligned) ----
  char* ws = (char*)d_ws;
  const size_t offCnt  = 0;                         // barrier counter (256 B slot)
  const size_t offH0   = 256;                       // 2 x [64,512] bf16 = 128 KB
  const size_t offH1   = offH0 + 131072;            // 128 KB
  const size_t offYp   = offH1 + 131072;            // 16*64 f32 = 4 KB
  const size_t offWe0  = offYp + 4096;              // 512*2048 bf16  = 2 MB
  const size_t offWe1  = offWe0 + 2097152;          // 1024*2048 bf16 = 4 MB
  const size_t offWd0  = offWe1 + 4194304;          // 2 MB
  const size_t offWd1  = offWd0 + 2097152;          // 4 MB

  int*            cnt  = (int*)(ws + offCnt);
  unsigned short* H0   = (unsigned short*)(ws + offH0);
  unsigned short* H1   = (unsigned short*)(ws + offH1);
  float*          yp   = (float*)(ws + offYp);
  unsigned short* We0  = (unsigned short*)(ws + offWe0);
  unsigned short* We1  = (unsigned short*)(ws + offWe1);
  unsigned short* Wd0  = (unsigned short*)(ws + offWd0);
  unsigned short* Wd1  = (unsigned short*)(ws + offWd1);

  // 1) zero the mutable region: cnt + H0 + H1 + yp = 266496 B = 16656 uint4
  ws_zero<<<(16656 + 255) / 256, 256, 0, stream>>>((uint4*)ws, 16656);

  // 2) pack f32 weights -> bf16 WMMA-fragment layout (stays hot in L2)
  pack_weights<<<8192, 256, 0, stream>>>(eWhh0, eWih1, eWhh1, dWhh0, dWih1, dWhh1,
                                         We0, We1, Wd0, Wd1);

  // 3) persistent recurrent kernel: 16 WGs (one 32-unit hidden slice each)
  lstm_seq2seq<<<NWG, TPB, 0, stream>>>(x,
                                        ebih0, ebhh0, eWih0,
                                        ebih1, ebhh1,
                                        dbih0, dbhh0, dWih0,
                                        dbih1, dbhh1,
                                        fcW, fcb,
                                        We0, We1, Wd0, Wd1,
                                        H0, H1, yp, cnt,
                                        (float*)d_out);
}
